// MVFAB_75943611728061
// MI455X (gfx1250) — matrix operand order repaired and checked by reading, not compiled, not run on hardware
//
#include <hip/hip_runtime.h>

// ---------------- problem constants ----------------
#define BB   2
#define CC   256
#define HH   64
#define WWD  64
#define NSP  4096         // H*W
#define MIPC 8
#define CQC  32
#define EPSV 1e-5f

typedef _Float16 h16;
typedef __attribute__((ext_vector_type(16))) _Float16 v16h;
typedef __attribute__((ext_vector_type(8)))  float    v8f;
typedef unsigned int u32x4 __attribute__((ext_vector_type(4)));
typedef int          i32x4 __attribute__((ext_vector_type(4)));
typedef int          i32x8 __attribute__((ext_vector_type(8)));

union AF { v16h v; h16 h[16]; };
union CF { v8f  v; float f[8]; };

__device__ inline void zero8(CF& c) {
#pragma unroll
  for (int i = 0; i < 8; i++) c.f[i] = 0.f;
}

// A-fragment K index for 16-bit 16x32 A (contiguous order, matches B fill)
__device__ inline int a_kidx(int e, int hs) {
  return hs * 16 + e;
}

// ---- Tensor Data Mover: 2D tile load Global->LDS (CDNA5 D#, ISA ch.8) ----
// tile_w/tile_h in elements (data_size = 2 bytes), stride = row stride in elems.
// padc_int / padc_amt: D# pad_interval / pad_amount codes (LDS row padding).
__device__ inline void tdm_load_2d(unsigned lds_off, const void* gptr,
                                   unsigned tile_w, unsigned tile_h,
                                   unsigned stride_elems,
                                   unsigned padc_int, unsigned padc_amt) {
  unsigned long long ga = (unsigned long long)(uintptr_t)gptr;
  u32x4 g0;
  g0[0] = 1u;                                        // count=1, user descriptor
  g0[1] = lds_off;                                   // lds_addr [63:32]
  g0[2] = (unsigned)(ga & 0xffffffffu);              // global_addr[31:0]
  g0[3] = (unsigned)((ga >> 32) & 0x01ffffffu) | (2u << 30);  // addr[56:32] | type=2
  i32x8 g1;
  g1[0] = (int)((1u << 16) |                         // data_size = 2 bytes
                (1u << 20) |                         // pad_enable
                (padc_int << 22) | (padc_amt << 25));
  g1[1] = (int)((stride_elems & 0xffffu) << 16);     // tensor_dim0[15:0] @bit48
  g1[2] = (int)(((stride_elems >> 16) & 0xffffu) |   // tensor_dim0[31:16]
                ((tile_h & 0xffffu) << 16));         // tensor_dim1[15:0]
  g1[3] = (int)(((tile_h >> 16) & 0xffffu) |         // tensor_dim1[31:16]
                ((tile_w & 0xffffu) << 16));         // tile_dim0
  g1[4] = (int)(tile_h & 0xffffu);                   // tile_dim1 (tile_dim2 = 0)
  g1[5] = (int)stride_elems;                         // tensor_dim0_stride[31:0]
  g1[6] = 0;                                         // stride hi, dim1_stride lo
  g1[7] = 0;
  i32x4 z4; z4[0] = 0; z4[1] = 0; z4[2] = 0; z4[3] = 0;
#if defined(__clang_major__) && (__clang_major__ >= 23)
  i32x8 z8;
#pragma unroll
  for (int i = 0; i < 8; i++) z8[i] = 0;
  __builtin_amdgcn_tensor_load_to_lds(g0, g1, z4, z4, z8, 0);
#else
  __builtin_amdgcn_tensor_load_to_lds(g0, g1, z4, z4, 0);
#endif
}

__device__ inline unsigned lds_offset(const void* p) {
  // generic LDS address = {SHARED_BASE, offset[31:0]} -> low 32 bits are the offset
  return (unsigned)(uintptr_t)p;
}

// ================= CoordAtt =================
__global__ void k_pool(const float* __restrict__ x, float* __restrict__ yp) {
  int bc = blockIdx.x;          // B*C
  int t  = threadIdx.x;         // 0..63
  const float* px = x + (size_t)bc * NSP;
  float sh = 0.f, sw = 0.f;
  for (int i = 0; i < 64; i++) { sh += px[t * 64 + i]; sw += px[i * 64 + t]; }
  yp[(size_t)bc * 128 + t]      = sh * (1.f / 64.f);
  yp[(size_t)bc * 128 + 64 + t] = sw * (1.f / 64.f);
}

__global__ void k_mid(const float* __restrict__ yp, const float* __restrict__ w1,
                      const float* __restrict__ b1, const float* __restrict__ bn,
                      float* __restrict__ ym) {
  int idx = blockIdx.x * blockDim.x + threadIdx.x;  // B*MIP*128
  if (idx >= BB * MIPC * 128) return;
  int p = idx & 127, o = (idx >> 7) & 7, b = idx >> 10;
  float acc = 0.f;
  for (int c = 0; c < CC; c++)
    acc += w1[o * CC + c] * yp[((size_t)b * CC + c) * 128 + p];
  acc += b1[o];
  float g = bn[o], be = bn[MIPC + o], mm = bn[2 * MIPC + o], vv = bn[3 * MIPC + o];
  float s = g * rsqrtf(vv + EPSV);
  float y = acc * s + (be - mm * s);
  ym[((size_t)b * MIPC + o) * 128 + p] =
      y * fminf(fmaxf(y + 3.f, 0.f), 6.f) * (1.f / 6.f);
}

__global__ void k_attnvec(const float* __restrict__ ym,
                          const float* __restrict__ wh, const float* __restrict__ bh,
                          const float* __restrict__ ww, const float* __restrict__ bw,
                          float* __restrict__ av) {
  int idx = blockIdx.x * blockDim.x + threadIdx.x;  // B*C*128
  if (idx >= BB * CC * 128) return;
  int p = idx & 127, c = (idx >> 7) & 255, b = idx >> 15;
  const float* yb = ym + (size_t)b * MIPC * 128;
  float acc;
  if (p < 64) {
    acc = bh[c];
    for (int o = 0; o < MIPC; o++) acc += wh[c * MIPC + o] * yb[o * 128 + p];
  } else {
    acc = bw[c];
    for (int o = 0; o < MIPC; o++) acc += ww[c * MIPC + o] * yb[o * 128 + p];
  }
  av[((size_t)b * CC + c) * 128 + p] = 1.f / (1.f + __expf(-acc));
}

__global__ void k_apply(const float* __restrict__ x, const float* __restrict__ av,
                        h16* __restrict__ rh) {
  long idx = (long)blockIdx.x * blockDim.x + threadIdx.x;  // B*C*N
  if (idx >= (long)BB * CC * NSP) return;
  int n = (int)(idx & (NSP - 1));
  int c = (int)((idx >> 12) & 255);
  int b = (int)(idx >> 20);
  int y = n >> 6, xx = n & 63;
  const float* a = av + ((size_t)b * CC + c) * 128;
  rh[idx] = (h16)(x[idx] * a[64 + xx] * a[y]);
}

// ================= helpers =================
__global__ void k_f2h(const float* __restrict__ s, h16* __restrict__ d, long n) {
  long i = (long)blockIdx.x * blockDim.x + threadIdx.x;
  if (i < n) d[i] = (h16)s[i];
}

__global__ void k_im2col(const h16* __restrict__ in, h16* __restrict__ col) {
  long idx = (long)blockIdx.x * blockDim.x + threadIdx.x;
  const long total = (long)BB * CC * 9 * NSP;
  if (idx >= total) return;
  int n  = (int)(idx & (NSP - 1));
  long r = idx >> 12;
  int k9 = (int)(r % 9);
  long r2 = r / 9;
  int c = (int)(r2 & 255);
  int b = (int)(r2 >> 8);
  int y = n >> 6, x = n & 63;
  int ky = k9 / 3 - 1, kx = k9 - (k9 / 3) * 3 - 1;
  int yy = y + ky, xx = x + kx;
  h16 v = (h16)0.f;
  if (yy >= 0 && yy < 64 && xx >= 0 && xx < 64)
    v = in[((size_t)b * CC + c) * NSP + yy * 64 + xx];
  col[((size_t)b * 2304 + (size_t)c * 9 + k9) * NSP + n] = v;
}

__global__ void k_poolcat(const float* __restrict__ x1, h16* __restrict__ cat,
                          int k, int sec) {
  long idx = (long)blockIdx.x * blockDim.x + threadIdx.x;  // B*C*N
  if (idx >= (long)BB * CC * NSP) return;
  int n = (int)(idx & (NSP - 1));
  int c = (int)((idx >> 12) & 255);
  int b = (int)(idx >> 20);
  int y = n >> 6, x = n & 63, p = k >> 1;
  const float* src = x1 + ((size_t)b * CC + c) * NSP;
  float m = -3.0e38f;
  for (int dy = -p; dy <= p; dy++) {
    int yy = y + dy; if (yy < 0 || yy >= 64) continue;
    for (int dx = -p; dx <= p; dx++) {
      int xx = x + dx; if (xx < 0 || xx >= 64) continue;
      m = fmaxf(m, src[yy * 64 + xx]);
    }
  }
  cat[((size_t)b * 1024 + sec + c) * NSP + n] = (h16)m;
}

// ================= WMMA GEMM (NN) with TDM operand staging =================
// D[o,n] = act(sum_k A[o,k] X[k,n]); mode: 0 linear, 1 BN+SiLU, 2 +bias
__global__ __launch_bounds__(256) void k_gemm(
    const h16* __restrict__ A, long AbS,
    const h16* __restrict__ X, long XbS,
    float* __restrict__ f32o, long f32S,
    h16* __restrict__ f16o, long f16S,
    const float* __restrict__ bn, const float* __restrict__ bias,
    int O, int K, int mode) {
  __shared__ h16 a_lds[32 * 40];   // 32x32 tile, 8-half row pad (TDM pad)
  __shared__ h16 b_lds[32 * 72];   // 32x64 tile, 8-half row pad (TDM pad)
  const int tid = threadIdx.x;
  const int lane = tid & 31, w = tid >> 5;
  const int hs = lane >> 4, l16 = lane & 15;
  const int mt = w >> 2, nt = w & 3;            // 2x4 wave grid -> 32x64 tile
  const int nBase = blockIdx.x * 64;
  const int oBase = blockIdx.y * 32;
  const int b = blockIdx.z;
  const h16* Ab = A + (size_t)b * AbS;
  const h16* Xb = X + (size_t)b * XbS;

  const unsigned a_off = lds_offset(a_lds);
  const unsigned b_off = lds_offset(b_lds);

  CF acc; zero8(acc);
#pragma unroll 1
  for (int k0 = 0; k0 < K; k0 += 32) {
    __syncthreads();                    // previous tile fully consumed
    if (w == 0) {
      // A tile: 32 rows x 32 halfs, row=64B -> pad_interval code 3 (16 DW),
      // pad 16B -> pad_amount code 3 (4 DW) => LDS row stride 40 halfs.
      tdm_load_2d(a_off, &Ab[(size_t)oBase * K + k0], 32, 32, (unsigned)K, 3, 3);
      // B tile: 32 rows x 64 halfs, row=128B -> pad_interval code 4 (32 DW),
      // pad 16B -> pad_amount code 3 => LDS row stride 72 halfs.
      tdm_load_2d(b_off, &Xb[(size_t)k0 * NSP + nBase], 64, 32, NSP, 4, 3);
      __builtin_amdgcn_s_wait_tensorcnt(0);
    }
    if (k0 + 32 < K)
      __builtin_prefetch(&Xb[(size_t)(k0 + 32) * NSP + nBase + tid * 16], 0, 1);
    __syncthreads();                    // tile visible to all waves
    AF af, bf;
#pragma unroll
    for (int e = 0; e < 16; e++) {
      af.h[e] = a_lds[(mt * 16 + l16) * 40 + a_kidx(e, hs)];
      bf.h[e] = b_lds[(hs * 16 + e) * 72 + nt * 16 + l16];
    }
    acc.v = __builtin_amdgcn_wmma_f32_16x16x32_f16(
        false, af.v, false, bf.v, (short)0, acc.v, false, false);
  }
  const int n_g = nBase + nt * 16 + l16;
#pragma unroll
  for (int v = 0; v < 8; v++) {
    int o = oBase + mt * 16 + v + 8 * hs;
    float val = acc.f[v];
    if (mode == 1) {
      float g = bn[o], be = bn[O + o], mm = bn[2 * O + o], vv = bn[3 * O + o];
      float s = g * rsqrtf(vv + EPSV);
      val = val * s + (be - mm * s);
      val = val / (1.f + __expf(-val));        // SiLU
    } else if (mode == 2) {
      val += bias[o];
    }
    if (f32o) f32o[(size_t)b * f32S + (size_t)o * NSP + n_g] = val;
    if (f16o) f16o[(size_t)b * f16S + (size_t)o * NSP + n_g] = (h16)val;
  }
}

// ================= WMMA GEMM (NT): E[m,d] = sum_n F[m,n] F[d,n] =================
__global__ __launch_bounds__(256) void k_gemm_nt(const h16* __restrict__ F,
                                                 float* __restrict__ Eo) {
  __shared__ h16 a_lds[32 * 40];
  __shared__ h16 b_lds[32 * 72];
  const int tid = threadIdx.x;
  const int lane = tid & 31, w = tid >> 5;
  const int hs = lane >> 4, l16 = lane & 15;
  const int mt = w >> 2, nt = w & 3;
  const int dBase = blockIdx.x * 64;
  const int mBase = blockIdx.y * 32;
  const int b = blockIdx.z;
  const h16* Fb = F + (size_t)b * CC * NSP;

  const int arow = tid >> 3, acol = (tid & 7) * 4;
  const int bc = tid >> 2, bk = (tid & 3) * 8;   // col 0..63, k chunk

  CF acc; zero8(acc);
#pragma unroll 1
  for (int k0 = 0; k0 < NSP; k0 += 32) {
    __syncthreads();
    *(uint2*)&a_lds[arow * 40 + acol] =
        *(const uint2*)&Fb[(size_t)(mBase + arow) * NSP + k0 + acol];
    uint4 d4 = *(const uint4*)&Fb[(size_t)(dBase + bc) * NSP + k0 + bk];
    const h16* hp = (const h16*)&d4;
#pragma unroll
    for (int j = 0; j < 8; j++) b_lds[(bk + j) * 72 + bc] = hp[j];
    __syncthreads();
    AF af, bf;
#pragma unroll
    for (int e = 0; e < 16; e++) {
      af.h[e] = a_lds[(mt * 16 + l16) * 40 + a_kidx(e, hs)];
      bf.h[e] = b_lds[(hs * 16 + e) * 72 + nt * 16 + l16];
    }
    acc.v = __builtin_amdgcn_wmma_f32_16x16x32_f16(
        false, af.v, false, bf.v, (short)0, acc.v, false, false);
  }
#pragma unroll
  for (int v = 0; v < 8; v++) {
    int m = mBase + mt * 16 + v + 8 * hs;
    int d = dBase + nt * 16 + l16;
    Eo[((size_t)b * CC + m) * CC + d] = acc.f[v];
  }
}

// ================= CAM softmax: attn = softmax(rowmax(e) - e) =================
__global__ void k_camsm(const float* __restrict__ e, h16* __restrict__ attn) {
  __shared__ float red[256];
  int bc = blockIdx.x;         // b*C + c
  int d = threadIdx.x;
  float ev = e[(size_t)bc * CC + d];
  red[d] = ev; __syncthreads();
  for (int s = 128; s > 0; s >>= 1) { if (d < s) red[d] = fmaxf(red[d], red[d + s]); __syncthreads(); }
  float emax = red[0]; __syncthreads();
  red[d] = ev; __syncthreads();
  for (int s = 128; s > 0; s >>= 1) { if (d < s) red[d] = fminf(red[d], red[d + s]); __syncthreads(); }
  float emin = red[0]; __syncthreads();
  float p = __expf((emax - ev) - (emax - emin));
  red[d] = p; __syncthreads();
  for (int s = 128; s > 0; s >>= 1) { if (d < s) red[d] += red[d + s]; __syncthreads(); }
  attn[(size_t)bc * CC + d] = (h16)(p / red[0]);
}

// ================= PAM flash attention (async-to-LDS K staging) =================
// pam[b][c][n] = sum_m softmax_m(q^T k)[n,m] * v[c,m]
__global__ __launch_bounds__(256) void k_pam(const h16* __restrict__ qh,
                                             const h16* __restrict__ kh,
                                             const h16* __restrict__ vh,
                                             float* __restrict__ pam) {
  __shared__ h16 k_lds[32 * 36];        // [c 0..31][key 0..31]
  __shared__ h16 v_lds[32 * 264];       // [key 0..31][c 0..255]
  __shared__ h16 p_lds[8 * 16 * 34];    // per-wave [query 16][key 32]
  const int tid = threadIdx.x;
  const int lane = tid & 31, w = tid >> 5;
  const int hs = lane >> 4, l16 = lane & 15;
  const int b = blockIdx.y;
  const int qBase = blockIdx.x * 128 + w * 16;
  const h16* qb = qh + (size_t)b * CQC * NSP;
  const h16* kb = kh + (size_t)b * CQC * NSP;
  const h16* vb = vh + (size_t)b * CC * NSP;

  AF qa;
#pragma unroll
  for (int e = 0; e < 16; e++)
    qa.h[e] = qb[(size_t)a_kidx(e, hs) * NSP + qBase + l16];

  CF acc[16];
#pragma unroll
  for (int ct = 0; ct < 16; ct++) zero8(acc[ct]);
  float mrow[8], lrow[8];
#pragma unroll
  for (int v = 0; v < 8; v++) { mrow[v] = -3.0e38f; lrow[v] = 0.f; }

  h16* pw = &p_lds[w * 16 * 34];
  const int kc = tid >> 3, kcol = (tid & 7) * 4;
  const int vc = tid >> 2, vkey8 = (tid & 3) * 8;
  const unsigned kdst = lds_offset(&k_lds[kc * 36 + kcol]);

#pragma unroll 1
  for (int m0 = 0; m0 < NSP; m0 += 32) {
    __syncthreads();
    {  // K tile: per-lane async copy global -> LDS (ASYNCcnt path)
      const h16* kg = &kb[(size_t)kc * NSP + m0 + kcol];
      asm volatile("global_load_async_to_lds_b64 %0, %1, off"
                   :: "v"(kdst), "v"(kg) : "memory");
    }
#pragma unroll
    for (int j = 0; j < 4; j++) {       // V tile: transpose scatter (layout change)
      int c = vc + j * 64;
      uint4 d4 = *(const uint4*)&vb[(size_t)c * NSP + m0 + vkey8];
      const h16* hp = (const h16*)&d4;
#pragma unroll
      for (int i = 0; i < 8; i++) v_lds[(vkey8 + i) * 264 + c] = hp[i];
    }
    asm volatile("s_wait_asynccnt 0" ::: "memory");
    __syncthreads();

    AF kf0, kf1;
#pragma unroll
    for (int e = 0; e < 16; e++) {
      int ke = hs * 16 + e;
      kf0.h[e] = k_lds[ke * 36 + l16];
      kf1.h[e] = k_lds[ke * 36 + 16 + l16];
    }
    CF s0, s1; zero8(s0); zero8(s1);
    s0.v = __builtin_amdgcn_wmma_f32_16x16x32_f16(false, qa.v, false, kf0.v,
                                                  (short)0, s0.v, false, false);
    s1.v = __builtin_amdgcn_wmma_f32_16x16x32_f16(false, qa.v, false, kf1.v,
                                                  (short)0, s1.v, false, false);
    float scl[8];
#pragma unroll
    for (int v = 0; v < 8; v++) {
      float t = fmaxf(s0.f[v], s1.f[v]);
#pragma unroll
      for (int off = 1; off < 16; off <<= 1) t = fmaxf(t, __shfl_xor(t, off, 16));
      float mnew = fmaxf(mrow[v], t);
      scl[v] = __expf(mrow[v] - mnew);
      float p0 = __expf(s0.f[v] - mnew);
      float p1 = __expf(s1.f[v] - mnew);
      float ls = p0 + p1;
#pragma unroll
      for (int off = 1; off < 16; off <<= 1) ls += __shfl_xor(ls, off, 16);
      lrow[v] = lrow[v] * scl[v] + ls;
      mrow[v] = mnew;
      int row = v + 8 * hs;
      pw[row * 34 + l16]      = (h16)p0;
      pw[row * 34 + 16 + l16] = (h16)p1;
    }
#pragma unroll
    for (int ct = 0; ct < 16; ct++)
#pragma unroll
      for (int v = 0; v < 8; v++) acc[ct].f[v] *= scl[v];
    AF pf;
#pragma unroll
    for (int e = 0; e < 16; e++)
      pf.h[e] = pw[l16 * 34 + a_kidx(e, hs)];
#pragma unroll
    for (int ct = 0; ct < 16; ct++) {
      AF vf;
#pragma unroll
      for (int e = 0; e < 16; e++)
        vf.h[e] = v_lds[(hs * 16 + e) * 264 + ct * 16 + l16];
      acc[ct].v = __builtin_amdgcn_wmma_f32_16x16x32_f16(
          false, pf.v, false, vf.v, (short)0, acc[ct].v, false, false);
    }
  }
#pragma unroll
  for (int ct = 0; ct < 16; ct++) {
    int c = ct * 16 + l16;
#pragma unroll
    for (int v = 0; v < 8; v++) {
      int qn = qBase + v + 8 * hs;
      pam[((size_t)b * CC + c) * NSP + qn] = acc[ct].f[v] / lrow[v];
    }
  }
}

// out = 3*res + g_pa*pam + g_ca*cam
__global__ void k_final(const float* __restrict__ res, const float* __restrict__ pam,
                        const float* __restrict__ cam, const float* __restrict__ gpa,
                        const float* __restrict__ gca, float* __restrict__ out) {
  long idx = (long)blockIdx.x * blockDim.x + threadIdx.x;
  if (idx >= (long)BB * CC * NSP) return;
  out[idx] = 3.f * res[idx] + gpa[0] * pam[idx] + gca[0] * cam[idx];
}

// ================= host launcher =================
extern "C" void kernel_launch(void* const* d_in, const int* in_sizes, int n_in,
                              void* d_out, int out_size, void* d_ws, size_t ws_size,
                              hipStream_t stream) {
  const float* x      = (const float*)d_in[0];
  const float* ca_w1  = (const float*)d_in[1];
  const float* ca_b1  = (const float*)d_in[2];
  const float* ca_bn  = (const float*)d_in[3];
  const float* ca_wh  = (const float*)d_in[4];
  const float* ca_bh  = (const float*)d_in[5];
  const float* ca_ww  = (const float*)d_in[6];
  const float* ca_bw  = (const float*)d_in[7];
  const float* s1_w = (const float*)d_in[8];  const float* s1_bn = (const float*)d_in[9];
  const float* s2_w = (const float*)d_in[10]; const float* s2_bn = (const float*)d_in[11];
  const float* s3_w = (const float*)d_in[12]; const float* s3_bn = (const float*)d_in[13];
  const float* s4_w = (const float*)d_in[14]; const float* s4_bn = (const float*)d_in[15];
  const float* s5_w = (const float*)d_in[16]; const float* s5_bn = (const float*)d_in[17];
  const float* s6_w = (const float*)d_in[18]; const float* s6_bn = (const float*)d_in[19];
  const float* s7_w = (const float*)d_in[20]; const float* s7_bn = (const float*)d_in[21];
  const float* pq_w = (const float*)d_in[22]; const float* pq_b = (const float*)d_in[23];
  const float* pk_w = (const float*)d_in[24]; const float* pk_b = (const float*)d_in[25];
  const float* pv_w = (const float*)d_in[26]; const float* pv_b = (const float*)d_in[27];
  const float* g_pa = (const float*)d_in[28]; const float* g_ca = (const float*)d_in[29];
  float* out = (float*)d_out;

  char* ws = (char*)d_ws;
  size_t off = 0;
  auto alloc = [&](size_t bytes) -> void* {
    off = (off + 255) & ~(size_t)255;
    void* p = ws + off;
    off += bytes;
    return p;
  };
  const size_t BCN = (size_t)BB * CC * NSP;

  float* yp    = (float*)alloc((size_t)BB * CC * 128 * 4);
  float* ym    = (float*)alloc((size_t)BB * MIPC * 128 * 4);
  float* av    = (float*)alloc((size_t)BB * CC * 128 * 4);
  h16*   r_h   = (h16*)  alloc(BCN * 2);
  h16*   w1h   = (h16*)  alloc((size_t)CC * CC * 2);
  h16*   w2h   = (h16*)  alloc((size_t)CC * CC * 2);
  h16*   w3h   = (h16*)  alloc((size_t)CC * 2304 * 2);
  h16*   w4h   = (h16*)  alloc((size_t)CC * CC * 2);
  h16*   w5h   = (h16*)  alloc((size_t)CC * 1024 * 2);
  h16*   w6h   = (h16*)  alloc((size_t)CC * 2304 * 2);
  h16*   w7h   = (h16*)  alloc((size_t)CC * 512 * 2);
  h16*   wqh   = (h16*)  alloc((size_t)CQC * CC * 2);
  h16*   wkh   = (h16*)  alloc((size_t)CQC * CC * 2);
  h16*   wvh   = (h16*)  alloc((size_t)CC * CC * 2);
  h16*   act0  = (h16*)  alloc(BCN * 2);
  h16*   act1  = (h16*)  alloc(BCN * 2);
  h16*   col_h = (h16*)  alloc((size_t)BB * 2304 * NSP * 2);
  float* x1_f  = (float*)alloc(BCN * 4);
  h16*   cat_h = (h16*)  alloc((size_t)BB * 1024 * NSP * 2);
  h16*   y12_h = (h16*)  alloc((size_t)BB * 512 * NSP * 2);
  float* res_f = (float*)alloc(BCN * 4);
  h16*   res_h = (h16*)  alloc(BCN * 2);
  h16*   q_h   = (h16*)  alloc((size_t)BB * CQC * NSP * 2);
  h16*   k_hb  = (h16*)  alloc((size_t)BB * CQC * NSP * 2);
  h16*   v_hb  = (h16*)  alloc(BCN * 2);
  float* pam_f = (float*)alloc(BCN * 4);
  float* e_f   = (float*)alloc((size_t)BB * CC * CC * 4);
  h16*   atc_h = (h16*)  alloc((size_t)BB * CC * CC * 2);
  float* cam_f = (float*)alloc(BCN * 4);
  (void)ws_size; (void)n_in; (void)in_sizes; (void)out_size;

  auto cvt = [&](const float* s, h16* d, long n) {
    k_f2h<<<dim3((unsigned)((n + 255) / 256)), dim3(256), 0, stream>>>(s, d, n);
  };
  cvt(s1_w, w1h, (long)CC * CC);
  cvt(s2_w, w2h, (long)CC * CC);
  cvt(s3_w, w3h, (long)CC * 2304);
  cvt(s4_w, w4h, (long)CC * CC);
  cvt(s5_w, w5h, (long)CC * 1024);
  cvt(s6_w, w6h, (long)CC * 2304);
  cvt(s7_w, w7h, (long)CC * 512);
  cvt(pq_w, wqh, (long)CQC * CC);
  cvt(pk_w, wkh, (long)CQC * CC);
  cvt(pv_w, wvh, (long)CC * CC);

  // ---- CoordAtt ----
  k_pool<<<dim3(BB * CC), dim3(64), 0, stream>>>(x, yp);
  k_mid<<<dim3(8), dim3(256), 0, stream>>>(yp, ca_w1, ca_b1, ca_bn, ym);
  k_attnvec<<<dim3(256), dim3(256), 0, stream>>>(ym, ca_wh, ca_bh, ca_ww, ca_bw, av);
  k_apply<<<dim3(8192), dim3(256), 0, stream>>>(x, av, r_h);

  const long CN = (long)CC * NSP;
  dim3 blk(256);
  auto ggrid = [](int O) { return dim3(NSP / 64, O / 32, BB); };

  // ---- SPPCSPC ----
  k_gemm<<<ggrid(CC), blk, 0, stream>>>(w1h, 0, r_h, CN, nullptr, 0, act0, CN,
                                        s1_bn, nullptr, CC, CC, 1);
  k_im2col<<<dim3(73728), blk, 0, stream>>>(act0, col_h);
  k_gemm<<<ggrid(CC), blk, 0, stream>>>(w3h, 0, col_h, (long)2304 * NSP, nullptr, 0,
                                        act1, CN, s3_bn, nullptr, CC, 2304, 1);
  k_gemm<<<ggrid(CC), blk, 0, stream>>>(w4h, 0, act1, CN, x1_f, CN, nullptr, 0,
                                        s4_bn, nullptr, CC, CC, 1);
  k_poolcat<<<dim3(8192), blk, 0, stream>>>(x1_f, cat_h, 1, 0);
  k_poolcat<<<dim3(8192), blk, 0, stream>>>(x1_f, cat_h, 5, CC);
  k_poolcat<<<dim3(8192), blk, 0, stream>>>(x1_f, cat_h, 9, 2 * CC);
  k_poolcat<<<dim3(8192), blk, 0, stream>>>(x1_f, cat_h, 13, 3 * CC);
  k_gemm<<<ggrid(CC), blk, 0, stream>>>(w5h, 0, cat_h, (long)1024 * NSP, nullptr, 0,
                                        act0, CN, s5_bn, nullptr, CC, 1024, 1);
  k_im2col<<<dim3(73728), blk, 0, stream>>>(act0, col_h);
  k_gemm<<<ggrid(CC), blk, 0, stream>>>(w6h, 0, col_h, (long)2304 * NSP, nullptr, 0,
                                        y12_h, (long)512 * NSP, s6_bn, nullptr, CC, 2304, 1);
  k_gemm<<<ggrid(CC), blk, 0, stream>>>(w2h, 0, r_h, CN, nullptr, 0,
                                        y12_h + (size_t)CC * NSP, (long)512 * NSP,
                                        s2_bn, nullptr, CC, CC, 1);
  k_gemm<<<ggrid(CC), blk, 0, stream>>>(w7h, 0, y12_h, (long)512 * NSP, res_f, CN,
                                        res_h, CN, s7_bn, nullptr, CC, 512, 1);

  // ---- PAM ----
  k_gemm<<<ggrid(CQC), blk, 0, stream>>>(wqh, 0, res_h, CN, nullptr, 0, q_h,
                                         (long)CQC * NSP, nullptr, pq_b, CQC, CC, 2);
  k_gemm<<<ggrid(CQC), blk, 0, stream>>>(wkh, 0, res_h, CN, nullptr, 0, k_hb,
                                         (long)CQC * NSP, nullptr, pk_b, CQC, CC, 2);
  k_gemm<<<ggrid(CC), blk, 0, stream>>>(wvh, 0, res_h, CN, nullptr, 0, v_hb, CN,
                                        nullptr, pv_b, CC, CC, 2);
  k_pam<<<dim3(NSP / 128, BB), blk, 0, stream>>>(q_h, k_hb, v_hb, pam_f);

  // ---- CAM ----
  k_gemm_nt<<<dim3(CC / 64, CC / 32, BB), blk, 0, stream>>>(res_h, e_f);
  k_camsm<<<dim3(BB * CC), blk, 0, stream>>>(e_f, atc_h);
  k_gemm<<<ggrid(CC), blk, 0, stream>>>(atc_h, (long)CC * CC, res_h, CN, cam_f, CN,
                                        nullptr, 0, nullptr, nullptr, CC, CC, 0);

  // ---- combine ----
  k_final<<<dim3(8192), blk, 0, stream>>>(res_f, pam_f, cam_f, g_pa, g_ca, out);
}
